// MHA_42880953483598
// MI455X (gfx1250) — matrix operand order repaired and checked by reading, not compiled, not run on hardware
//
#include <hip/hip_runtime.h>
#include <hip/hip_bf16.h>
#include <math.h>

#define S_LEN   4096
#define D_MODEL 768
#define N_HEADS 12
#define D_HEAD  64

typedef __attribute__((ext_vector_type(16))) __bf16        bf16x16;
typedef __attribute__((ext_vector_type(8)))  float         f32x8;
typedef __attribute__((ext_vector_type(4)))  unsigned int  u32x4;
typedef __attribute__((ext_vector_type(8)))  unsigned int  u32x8;

// round-to-nearest-even f32 -> bf16
static __device__ __forceinline__ unsigned short f32_to_bf16(float f) {
  unsigned u = __builtin_bit_cast(unsigned, f);
  unsigned r = u + 0x7FFFu + ((u >> 16) & 1u);
  return (unsigned short)(r >> 16);
}

// A fragment: 16x32 bf16 (MxK). Lane l holds row l%16.
// VGPR j : k = 16*half + 2j, 2j+1
// => per lane two contiguous 16B runs at k = 16*half and k = 16*half + 8.
static __device__ __forceinline__ bf16x16 load_frag_A(const unsigned short* s, int ld, int lane) {
  int row = lane & 15, h = lane >> 4;
  const u32x4* p0 = (const u32x4*)(s + (size_t)row * ld + 16 * h);
  const u32x4* p1 = (const u32x4*)(s + (size_t)row * ld + 16 * h + 8);
  u32x4 a = *p0, b = *p1;
  u32x8 v;
  v[0] = a[0]; v[1] = a[1]; v[2] = a[2]; v[3] = a[3];
  v[4] = b[0]; v[5] = b[1]; v[6] = b[2]; v[7] = b[3];
  return __builtin_bit_cast(bf16x16, v);
}

// B fragment: 32x16 bf16 (KxN), where B[k][n] = src[n*ld + k] (i.e. src holds
// columns of B contiguously; for W this is W[n][k] row-major directly).
// Lane l holds col l%16, k = 16*half + 0..15 => one contiguous 32B run.
static __device__ __forceinline__ bf16x16 load_frag_B(const unsigned short* s, int ld, int lane) {
  int n = lane & 15, h = lane >> 4;
  const u32x4* p = (const u32x4*)(s + (size_t)n * ld + 16 * h);
  u32x4 a = p[0], b = p[1];
  u32x8 v;
  v[0] = a[0]; v[1] = a[1]; v[2] = a[2]; v[3] = a[3];
  v[4] = b[0]; v[5] = b[1]; v[6] = b[2]; v[7] = b[3];
  return __builtin_bit_cast(bf16x16, v);
}

static __device__ __forceinline__ f32x8 wmma_bf16(bf16x16 a, bf16x16 b, f32x8 c) {
  return __builtin_amdgcn_wmma_f32_16x16x32_bf16(false, a, false, b, (short)0, c, false, false);
}

// ---------------- k0: fp32 -> bf16 cast ----------------
__global__ void k0_convert(const float* __restrict__ src, unsigned short* __restrict__ dst, int n) {
  int i = blockIdx.x * blockDim.x + threadIdx.x;
  if (i < n) dst[i] = f32_to_bf16(src[i]);
}

// ---------------- k1: QKV projection + RoPE ----------------
// grid: (S/64, H, 3[q,k,v]); block: 128 (4 waves, each a 16-row block).
// Each wave: 16 rows x 64 cols (one head) of x @ W^T, K-loop over 768.
__global__ void k1_qkv_rope(const unsigned short* __restrict__ Xb,
                            const unsigned short* __restrict__ Wqkv,
                            unsigned short* __restrict__ Qh,
                            unsigned short* __restrict__ Kh,
                            unsigned short* __restrict__ Vt) {
  int lane = threadIdx.x & 31;
  int wid  = threadIdx.x >> 5;
  int sblk = blockIdx.x * 4 + wid;
  int head = blockIdx.y;
  int z    = blockIdx.z;
  const unsigned short* W = Wqkv + (size_t)z * D_MODEL * D_MODEL;
  int s0 = sblk * 16;
  int h = lane >> 4, c = lane & 15;

  f32x8 acc[4] = {};
  for (int kc = 0; kc < D_MODEL; kc += 32) {
    bf16x16 a = load_frag_A(Xb + (size_t)s0 * D_MODEL + kc, D_MODEL, lane);
#pragma unroll
    for (int t = 0; t < 4; ++t) {
      int n0 = head * D_HEAD + t * 16;
      bf16x16 b = load_frag_B(W + (size_t)n0 * D_MODEL + kc, D_MODEL, lane);
      acc[t] = wmma_bf16(a, b, acc[t]);
    }
  }

  if (z < 2) {
    // RoPE in f32 accumulators. C-layout: col n = t*16 + c, row s = s0 + r + 8h.
    // Pair (2i, 2i+1) = adjacent lanes (parity of c). angle = s * theta^(-2i/64).
#pragma unroll
    for (int t = 0; t < 4; ++t) {
      int n = t * 16 + c;
      float freq = expf(-(float)(n & ~1) * (9.210340372f / 64.0f)); // 10000^(-2i/64)
#pragma unroll
      for (int r = 0; r < 8; ++r) {
        int srow = s0 + r + 8 * h;
        float ang = (float)srow * freq;
        float sn, cs;
        __sincosf(ang, &sn, &cs);
        float v = acc[t][r];
        float other = __shfl_xor(v, 1, 32); // partner element of the pair
        acc[t][r] = (n & 1) ? (other * sn + v * cs)   // odd:  x1*sin + x2*cos
                            : (v * cs - other * sn);  // even: x1*cos - x2*sin
      }
    }
    unsigned short* O = (z == 0) ? Qh : Kh;
#pragma unroll
    for (int t = 0; t < 4; ++t)
#pragma unroll
      for (int r = 0; r < 8; ++r) {
        int srow = s0 + r + 8 * h;
        O[((size_t)head * S_LEN + srow) * D_HEAD + t * 16 + c] = f32_to_bf16(acc[t][r]);
      }
  } else {
    // V stored transposed: Vt[head][dim][s] so the P*V B-fragment is contiguous.
#pragma unroll
    for (int t = 0; t < 4; ++t)
#pragma unroll
      for (int r = 0; r < 8; ++r) {
        int srow = s0 + r + 8 * h;
        Vt[((size_t)head * D_HEAD + t * 16 + c) * S_LEN + srow] = f32_to_bf16(acc[t][r]);
      }
  }
}

// ---------------- k2: causal flash attention ----------------
// grid: (S/64, H); block 128 (4 waves, each owns a 16-row query block).
__global__ void k2_flash(const unsigned short* __restrict__ Qh,
                         const unsigned short* __restrict__ Kh,
                         const unsigned short* __restrict__ Vt,
                         unsigned short* __restrict__ Ocat) {
  __shared__ unsigned short ptile[4][16 * 32]; // 1KB per wave
  int lane = threadIdx.x & 31;
  int wid  = threadIdx.x >> 5;
  int qb   = blockIdx.x * 4 + wid;
  int head = blockIdx.y;
  int h = lane >> 4, c = lane & 15;

  const unsigned short* Q = Qh + (size_t)head * S_LEN * D_HEAD;
  const unsigned short* K = Kh + (size_t)head * S_LEN * D_HEAD;
  const unsigned short* V = Vt + (size_t)head * D_HEAD * S_LEN;
  unsigned short* pt = ptile[wid];

  // Q A-fragments (16 rows x 64 dims = two K=32 fragments), resident in VGPRs
  bf16x16 qa0 = load_frag_A(Q + (size_t)qb * 16 * D_HEAD,      D_HEAD, lane);
  bf16x16 qa1 = load_frag_A(Q + (size_t)qb * 16 * D_HEAD + 32, D_HEAD, lane);

  f32x8 o[4] = {};
  float mrow[8], lrow[8];
#pragma unroll
  for (int r = 0; r < 8; ++r) { mrow[r] = -__builtin_inff(); lrow[r] = 0.0f; }

  int ktmax = (qb * 16 + 15) >> 5; // last 32-key tile touching the causal diagonal
  for (int kt = 0; kt <= ktmax; ++kt) {
    // scores: 16 rows x 32 keys = two 16x16 f32 accs, each sum of 2 WMMAs over d_k
    f32x8 sc[2] = {};
#pragma unroll
    for (int u = 0; u < 2; ++u) {
      const unsigned short* kb = K + (size_t)(kt * 32 + u * 16) * D_HEAD;
      bf16x16 b0 = load_frag_B(kb,      D_HEAD, lane); // B[k][n] = K[key n][k]
      bf16x16 b1 = load_frag_B(kb + 32, D_HEAD, lane);
      sc[u] = wmma_bf16(qa0, b0, sc[u]);
      sc[u] = wmma_bf16(qa1, b1, sc[u]);
    }
#pragma unroll
    for (int r = 0; r < 8; ++r) {
      int srow = qb * 16 + r + 8 * h;
      int k0 = kt * 32 + c;
      float v0 = (k0      <= srow) ? sc[0][r] * 0.125f : -__builtin_inff();
      float v1 = (k0 + 16 <= srow) ? sc[1][r] * 0.125f : -__builtin_inff();
      float cand = fmaxf(v0, v1);
#pragma unroll
      for (int off = 1; off < 16; off <<= 1)
        cand = fmaxf(cand, __shfl_xor(cand, off, 32)); // row max across 16 lanes
      float mnew  = fmaxf(mrow[r], cand);               // finite: key 0 always valid
      float alpha = expf(mrow[r] - mnew);
      float p0 = expf(v0 - mnew);
      float p1 = expf(v1 - mnew);
      float rs = p0 + p1;
#pragma unroll
      for (int off = 1; off < 16; off <<= 1)
        rs += __shfl_xor(rs, off, 32);                  // row sum
      lrow[r] = lrow[r] * alpha + rs;
      mrow[r] = mnew;
#pragma unroll
      for (int t = 0; t < 4; ++t) o[t][r] *= alpha;     // rescale O
      int prow = r + 8 * h;                             // C-layout -> LDS row-major P
      pt[prow * 32 + c]      = f32_to_bf16(p0);
      pt[prow * 32 + 16 + c] = f32_to_bf16(p1);
    }
    asm volatile("s_wait_dscnt 0x0" ::: "memory");      // LDS RAW within the wave
    bf16x16 pa = load_frag_A(pt, 32, lane);             // P as 16x32 A-fragment
#pragma unroll
    for (int t = 0; t < 4; ++t) {
      // B[k][n] = V[key k][dim n] = Vt[dim][key] : contiguous per lane
      bf16x16 vb = load_frag_B(V + (size_t)(t * 16) * S_LEN + kt * 32, S_LEN, lane);
      o[t] = wmma_bf16(pa, vb, o[t]);
    }
  }

#pragma unroll
  for (int t = 0; t < 4; ++t)
#pragma unroll
    for (int r = 0; r < 8; ++r) {
      int srow = qb * 16 + r + 8 * h;
      int col  = head * D_HEAD + t * 16 + c;
      Ocat[(size_t)srow * D_MODEL + col] = f32_to_bf16(o[t][r] / lrow[r]);
    }
}

// ---------------- k3: output projection ----------------
// grid: (S/64, D/64); block 128. out = Ocat @ Wo^T (fp32 store).
__global__ void k3_oproj(const unsigned short* __restrict__ Ocat,
                         const unsigned short* __restrict__ Wob,
                         float* __restrict__ out) {
  int lane = threadIdx.x & 31;
  int wid  = threadIdx.x >> 5;
  int s0   = (blockIdx.x * 4 + wid) * 16;
  int nb   = blockIdx.y * 64;
  int h = lane >> 4, c = lane & 15;

  f32x8 acc[4] = {};
  for (int kc = 0; kc < D_MODEL; kc += 32) {
    bf16x16 a = load_frag_A(Ocat + (size_t)s0 * D_MODEL + kc, D_MODEL, lane);
#pragma unroll
    for (int t = 0; t < 4; ++t) {
      bf16x16 b = load_frag_B(Wob + (size_t)(nb + t * 16) * D_MODEL + kc, D_MODEL, lane);
      acc[t] = wmma_bf16(a, b, acc[t]);
    }
  }
#pragma unroll
  for (int t = 0; t < 4; ++t)
#pragma unroll
    for (int r = 0; r < 8; ++r)
      out[(size_t)(s0 + r + 8 * h) * D_MODEL + nb + t * 16 + c] = acc[t][r];
}

extern "C" void kernel_launch(void* const* d_in, const int* in_sizes, int n_in,
                              void* d_out, int out_size, void* d_ws, size_t ws_size,
                              hipStream_t stream) {
  (void)in_sizes; (void)n_in; (void)out_size; (void)ws_size;
  const float* x  = (const float*)d_in[0];
  const float* Wq = (const float*)d_in[1];
  const float* Wk = (const float*)d_in[2];
  const float* Wv = (const float*)d_in[3];
  const float* Wo = (const float*)d_in[4];
  float* out = (float*)d_out;

  const size_t NX = (size_t)S_LEN * D_MODEL;      // 3,145,728
  const size_t NW = (size_t)D_MODEL * D_MODEL;    //   589,824
  const size_t NH = (size_t)N_HEADS * S_LEN * D_HEAD; // 3,145,728

  unsigned short* Xb   = (unsigned short*)d_ws;
  unsigned short* Wqkv = Xb + NX;        // Wq,Wk,Wv bf16, contiguous
  unsigned short* Wqb  = Wqkv;
  unsigned short* Wkb  = Wqb + NW;
  unsigned short* Wvb  = Wkb + NW;
  unsigned short* Wob  = Wvb + NW;
  unsigned short* Qh   = Wob + NW;       // [H][S][64]
  unsigned short* Kh   = Qh + NH;        // [H][S][64]
  unsigned short* Vt   = Kh + NH;        // [H][64][S]
  unsigned short* Ocat = Vt + NH;        // [S][768]
  // total ws: ~36.2 MB

  k0_convert<<<(int)((NX + 255) / 256), 256, 0, stream>>>(x,  Xb,  (int)NX);
  k0_convert<<<(int)((NW + 255) / 256), 256, 0, stream>>>(Wq, Wqb, (int)NW);
  k0_convert<<<(int)((NW + 255) / 256), 256, 0, stream>>>(Wk, Wkb, (int)NW);
  k0_convert<<<(int)((NW + 255) / 256), 256, 0, stream>>>(Wv, Wvb, (int)NW);
  k0_convert<<<(int)((NW + 255) / 256), 256, 0, stream>>>(Wo, Wob, (int)NW);

  dim3 g1(S_LEN / 64, N_HEADS, 3);
  k1_qkv_rope<<<g1, 128, 0, stream>>>(Xb, Wqkv, Qh, Kh, Vt);

  dim3 g2(S_LEN / 64, N_HEADS);
  k2_flash<<<g2, 128, 0, stream>>>(Qh, Kh, Vt, Ocat);

  dim3 g3(S_LEN / 64, D_MODEL / 64);
  k3_oproj<<<g3, 128, 0, stream>>>(Ocat, Wob, out);
}
